// CP2_17669495456475
// MI455X (gfx1250) — matrix operand order repaired and checked by reading, not compiled, not run on hardware
//
#include <hip/hip_runtime.h>

// ---------------------------------------------------------------------------
// cos_similar: (4, 1024, 32, 32) f32 ; b: (4, 32, 256, 256) f32 ;
// mask: (4, 1, 256, 256) f32 ; out: (4, 32, 256, 256) f32
// Per batch GEMM: M=1024 (p=(hs,ws)), K=1024 (l=(ph,pw)), N=8192 (c,ky,kx),
// bf16 inputs / f32 accumulate, then stride-8/kernel-16 deconv scatter-add
// (<=4 overlapping adds per output pixel) with f32 atomics.
// Workspace: Abt (2 MB bf16) + Wbt (16 MB bf16), reused across batches.
// ---------------------------------------------------------------------------

typedef __attribute__((ext_vector_type(16))) __bf16 bf16x16;
typedef __attribute__((ext_vector_type(8)))  __bf16 bf16x8;
typedef __attribute__((ext_vector_type(8)))  float  f32x8;

#define STRIDE 8
#define PDPAD  4
#define LDIM   1024      // K dimension (patches)
#define MDIM   1024      // M dimension (spatial positions)
#define NDIM   8192      // N dimension (c*ky*kx = 32*16*16)
#define IMG    256
#define CCH    32
#define KSTEPS (LDIM / 32)

// --- CDNA5 async global->LDS path (inline asm; ASYNCcnt wait via builtin) --
#if defined(__HIP_DEVICE_COMPILE__)
#if __has_builtin(__builtin_amdgcn_s_wait_asynccnt)
#define USE_ASYNC_LDS 1
#endif
#endif
#ifndef USE_ASYNC_LDS
#define USE_ASYNC_LDS 0
#endif

#if USE_ASYNC_LDS
#define WAIT_ASYNC(N) __builtin_amdgcn_s_wait_asynccnt(N)
#else
#define WAIT_ASYNC(N) ((void)0)
#endif

__device__ __forceinline__ void copy16_g2l(const unsigned short* g,
                                           unsigned short* l) {
#if USE_ASYNC_LDS
    // VDST = per-lane LDS byte address (low 32 bits of the generic shared
    // pointer ARE the LDS offset per the flat-aperture decode), VADDR = 64-bit
    // global address, no SADDR. Tracked by ASYNCcnt.
    unsigned loff = (unsigned)(size_t)l;
    asm volatile("global_load_async_to_lds_b128 %0, %1, off"
                 :
                 : "v"(loff), "v"(g)
                 : "memory");
#else
    *(uint4*)l = *(const uint4*)g;
#endif
}

__device__ __forceinline__ unsigned short f2bf_rne(float f) {
    union { float f; unsigned u; } v; v.f = f;
    unsigned r = v.u + 0x7FFFu + ((v.u >> 16) & 1u);   // round-to-nearest-even
    return (unsigned short)(r >> 16);
}

__global__ void zero_out(float* __restrict__ out, int n) {
    int i = blockIdx.x * blockDim.x + threadIdx.x;
    int stride = gridDim.x * blockDim.x;
    for (; i < n; i += stride) out[i] = 0.0f;
}

// Transpose-convert cos[b] (l, p) f32 -> Abt (p, l) bf16 via LDS 32x32 tiles
// so both global accesses are coalesced.
__global__ void prep_a(const float* __restrict__ cosb,
                       unsigned short* __restrict__ Abt) {
    __shared__ float t[32][33];
    const int tile = blockIdx.x;             // 1024 tiles
    const int p0 = (tile & 31) * 32;
    const int l0 = (tile >> 5) * 32;
    const int tx = threadIdx.x & 31;
    const int ty = threadIdx.x >> 5;         // 0..7
#pragma unroll
    for (int r = 0; r < 4; ++r) {
        int row = ty * 4 + r;
        t[row][tx] = cosb[(size_t)(l0 + row) * MDIM + p0 + tx];
    }
    __syncthreads();
#pragma unroll
    for (int r = 0; r < 4; ++r) {
        int row = ty * 4 + r;
        Abt[(size_t)(p0 + row) * LDIM + l0 + tx] = f2bf_rne(t[tx][row]);
    }
}

// Wbt[n][l] = b_pad[c, ph*8+ky, pw*8+kx] * (1 - m_pad[...]); replicate pad
// realized as index clamp. n = c*256 + ky*16 + kx ; l = ph*32 + pw.
__global__ void prep_w(const float* __restrict__ bimg,   // (32,256,256)
                       const float* __restrict__ mask,   // (256,256)
                       unsigned short* __restrict__ Wbt) {
    int idx = blockIdx.x * 256 + threadIdx.x;            // 0 .. 8M-1
    int n = idx >> 10, l = idx & 1023;
    int c  = n >> 8;
    int ky = (n >> 4) & 15;
    int kx = n & 15;
    int ph = l >> 5, pw = l & 31;
    int y = ph * STRIDE + ky - PDPAD;  y = min(max(y, 0), IMG - 1);
    int x = pw * STRIDE + kx - PDPAD;  x = min(max(x, 0), IMG - 1);
    float m = mask[y * IMG + x];
    Wbt[idx] = f2bf_rne(bimg[(c * IMG + y) * IMG + x] * (1.0f - m));
}

// 128x128 tile per 256-thread block; 8 wave32s in 4(M) x 2(N) grid; each wave
// owns 32x64 = 2x4 WMMA accumulators. Double-buffered LDS; async global->LDS
// staging overlaps the previous tile's WMMAs.
__launch_bounds__(256, 2)
__global__ void gemm_scatter(const unsigned short* __restrict__ Abt,  // (M,K) bf16
                             const unsigned short* __restrict__ Wbt,  // (N,K) bf16
                             float* __restrict__ out, int batch) {
    __shared__ unsigned short Alds[2][128][40];   // rows padded to 80B stride
    __shared__ unsigned short Blds[2][128][40];

    const int tid  = threadIdx.x;
    const int lane = tid & 31;
    const int wave = tid >> 5;
    const int wm   = wave & 3;     // wave row (32 M-rows each)
    const int wn   = wave >> 2;    // wave col (64 N-cols each)
    const int half = lane >> 4;    // lane half (VGPR layout split)
    const int lrow = lane & 15;

    const int m0 = blockIdx.y * 128;
    const int n0 = blockIdx.x * 128;

    // each thread stages rows (tid>>2) and (tid>>2)+64, one 16B chunk each,
    // for both A and B tiles: exactly 4 B128 copies per k-step.
    const int srow = tid >> 2;            // 0..63
    const int scol = (tid & 3) * 8;       // bf16 element offset 0,8,16,24

    auto stage = [&](int buf, int kk) {
        copy16_g2l(Abt + (size_t)(m0 + srow) * LDIM + kk + scol,
                   &Alds[buf][srow][scol]);
        copy16_g2l(Abt + (size_t)(m0 + srow + 64) * LDIM + kk + scol,
                   &Alds[buf][srow + 64][scol]);
        copy16_g2l(Wbt + (size_t)(n0 + srow) * LDIM + kk + scol,
                   &Blds[buf][srow][scol]);
        copy16_g2l(Wbt + (size_t)(n0 + srow + 64) * LDIM + kk + scol,
                   &Blds[buf][srow + 64][scol]);
    };

    f32x8 acc[2][4] = {};

    stage(0, 0);
    for (int kt = 0; kt < KSTEPS; ++kt) {
        const int buf = kt & 1;
        if (kt + 1 < KSTEPS) {
            stage(buf ^ 1, (kt + 1) * 32);   // prefetch next tile (async)
            WAIT_ASYNC(4);                   // current tile's 4 copies done
        } else {
            WAIT_ASYNC(0);
        }
        __syncthreads();

        // A 16x32 fragment: lanes 0-15 row M=lane; elems 0-7 = K(half*8..+7),
        // elems 8-15 = K(16+half*8..+7).
        bf16x16 afrag[2];
#pragma unroll
        for (int i = 0; i < 2; ++i) {
            int r = wm * 32 + i * 16 + lrow;
            bf16x8 lo = *(const bf16x8*)&Alds[buf][r][half * 8];
            bf16x8 hi = *(const bf16x8*)&Alds[buf][r][16 + half * 8];
            afrag[i] = __builtin_shufflevector(lo, hi,
                0,1,2,3,4,5,6,7,8,9,10,11,12,13,14,15);
        }
        // B 32x16 fragment: col N=lane%16; elems 0-7 = K(half*8..+7), elems 8-15 = K(16+half*8..+7).
        bf16x16 bfrag[4];
#pragma unroll
        for (int j = 0; j < 4; ++j) {
            int r = wn * 64 + j * 16 + lrow;
            bf16x8 lo = *(const bf16x8*)&Blds[buf][r][half * 8];
            bf16x8 hi = *(const bf16x8*)&Blds[buf][r][16 + half * 8];
            bfrag[j] = __builtin_shufflevector(lo, hi,
                0,1,2,3,4,5,6,7,8,9,10,11,12,13,14,15);
        }

#pragma unroll
        for (int i = 0; i < 2; ++i)
#pragma unroll
            for (int j = 0; j < 4; ++j)
                acc[i][j] = __builtin_amdgcn_wmma_f32_16x16x32_bf16(
                    false, afrag[i], false, bfrag[j],
                    (short)0, acc[i][j], false, false);
        __syncthreads();
    }

    // --- deconv scatter-add epilogue ---
    // Within a 16x16 tile: c, ky are constant (from n-tile base) and hs is
    // constant (m-tile is 16-aligned inside one 32-row band), so y and the
    // output row pointer are scalar per tile; only x varies per element.
#pragma unroll
    for (int i = 0; i < 2; ++i) {
        const int mbase  = m0 + wm * 32 + i * 16;   // 16-aligned
        const int hs     = mbase >> 5;              // uniform over tile
        const int wsbase = mbase & 31;              // 0 or 16
#pragma unroll
        for (int j = 0; j < 4; ++j) {
            const int nbase = n0 + wn * 64 + j * 16;
            const int c  = nbase >> 8;
            const int ky = (nbase >> 4) & 15;
            const int y  = hs * STRIDE + ky - PDPAD;
            if ((unsigned)y < IMG) {
                float* rowp = out +
                    (((size_t)batch * CCH + c) * IMG + y) * IMG;
#pragma unroll
                for (int r = 0; r < 8; ++r) {
                    int wsx = wsbase + half * 8 + r;            // ws
                    int x = wsx * STRIDE + lrow - PDPAD;        // kx = lrow
                    if ((unsigned)x < IMG)
                        __hip_atomic_fetch_add(rowp + x, acc[i][j][r],
                                               __ATOMIC_RELAXED,
                                               __HIP_MEMORY_SCOPE_AGENT);
                }
            }
        }
    }
}

extern "C" void kernel_launch(void* const* d_in, const int* in_sizes, int n_in,
                              void* d_out, int out_size, void* d_ws, size_t ws_size,
                              hipStream_t stream) {
    const float* cosp = (const float*)d_in[0];   // (4,1024,32,32)
    const float* bimg = (const float*)d_in[1];   // (4,32,256,256)
    const float* mask = (const float*)d_in[2];   // (4,1,256,256)
    float* out = (float*)d_out;                  // (4,32,256,256)

    unsigned short* Abt = (unsigned short*)d_ws;                 // 2 MB
    unsigned short* Wbt = Abt + (size_t)MDIM * LDIM;             // 16 MB
    (void)in_sizes; (void)n_in; (void)ws_size;

    zero_out<<<2048, 256, 0, stream>>>(out, out_size);

    for (int b = 0; b < 4; ++b) {
        prep_a<<<1024, 256, 0, stream>>>(cosp + (size_t)b * LDIM * MDIM, Abt);
        prep_w<<<(NDIM * LDIM) / 256, 256, 0, stream>>>(
            bimg + (size_t)b * CCH * IMG * IMG,
            mask + (size_t)b * IMG * IMG, Wbt);
        dim3 grid(NDIM / 128, MDIM / 128);       // 64 x 8 blocks
        gemm_scatter<<<grid, 256, 0, stream>>>(Abt, Wbt, out, b);
    }
}
